// GraphAttentionNetwork_56186762166720
// MI455X (gfx1250) — matrix operand order repaired and checked by reading, not compiled, not run on hardware
//
#include <hip/hip_runtime.h>
#include <hip/hip_bf16.h>

typedef __attribute__((ext_vector_type(16))) _Float16 v16h;
typedef __attribute__((ext_vector_type(8)))  float    v8f;

#define NN     4096
#define IN_NF  128
#define HID    256
#define ONF    256
#define HEADS  4
#define FDIM   64
#define SLOPE  0.2f

__device__ __forceinline__ float lrelu(float x) { return x >= 0.f ? x : SLOPE * x; }

// ---------------------------------------------------------------------------
// Kernel 1: g = nodes @ W_emb.T  (f32 + f16-transposed copy), s_l/s_r per head
// grid 4096 blocks x 256 threads (one block per node, one thread per channel)
// ---------------------------------------------------------------------------
__global__ void __launch_bounds__(256)
embed_kernel(const float* __restrict__ nodes, const float* __restrict__ W_emb,
             const float* __restrict__ W_att,
             float* __restrict__ g,       // [NN, 256]
             _Float16* __restrict__ gT,   // [256, NN]
             float* __restrict__ s_lT,    // [4, NN]
             float* __restrict__ s_rT)    // [4, NN]
{
    __shared__ float xrow[IN_NF];
    __shared__ float slp[ONF];
    __shared__ float srp[ONF];
    int i = blockIdx.x;
    int c = threadIdx.x;
    if (c < IN_NF) xrow[c] = nodes[i * IN_NF + c];
    __syncthreads();
    const float* w = W_emb + c * IN_NF;
    float acc = 0.f;
#pragma unroll 8
    for (int k = 0; k < IN_NF; ++k) acc += xrow[k] * w[k];
    g[(size_t)i * ONF + c] = acc;
    gT[(size_t)c * NN + i] = (_Float16)acc;
    int f = c & (FDIM - 1);
    slp[c] = acc * W_att[f];
    srp[c] = acc * W_att[FDIM + f];
    __syncthreads();
    if (c < HEADS) {
        float a = 0.f, b = 0.f;
        for (int f2 = 0; f2 < FDIM; ++f2) { a += slp[c * FDIM + f2]; b += srp[c * FDIM + f2]; }
        s_lT[c * NN + i] = a;
        s_rT[c * NN + i] = b;
    }
}

// ---------------------------------------------------------------------------
// Kernel 2: softmax denominators Z[i,h] = sum_j adj[i,j]*exp(lrelu(sl+sr))
// Branchless (adj in {0,1} is applied as a multiply). One wave per row.
// grid 512 x 256 (8 waves/block)
// ---------------------------------------------------------------------------
__global__ void __launch_bounds__(256)
zsum_kernel(const int* __restrict__ adj, const float* __restrict__ s_lT,
            const float* __restrict__ s_rT, float* __restrict__ ZinvT)
{
    int wave = threadIdx.x >> 5;
    int lane = threadIdx.x & 31;
    int i = blockIdx.x * 8 + wave;
    float sl[HEADS], accv[HEADS];
#pragma unroll
    for (int h = 0; h < HEADS; ++h) { sl[h] = s_lT[h * NN + i]; accv[h] = 0.f; }
    const int* arow = adj + (size_t)i * NN;
    for (int j = lane; j < NN; j += 32) {
        float m = (float)arow[j];     // 0.0 or 1.0, no branch
#pragma unroll
        for (int h = 0; h < HEADS; ++h)
            accv[h] += m * __expf(lrelu(sl[h] + s_rT[h * NN + j]));
    }
#pragma unroll
    for (int h = 0; h < HEADS; ++h) {
        float v = accv[h];
        for (int off = 16; off; off >>= 1) v += __shfl_down(v, off, 32);
        if (lane == 0) ZinvT[h * NN + i] = 1.f / v;
    }
}

// ---------------------------------------------------------------------------
// Kernel 3: WMMA aggregation. h_agg[i, h*64+f] = (1/Z) * sum_j P[i,j,h]*g[j,h,f]
// One block = 16 output rows; wave w = head w. K-loop over 4096 in steps of 32.
// A (16x32 f16) built branchlessly from adjacency + exp(lrelu), B from gT.
// grid 256 x 128
// ---------------------------------------------------------------------------
__global__ void __launch_bounds__(128)
agg_kernel(const int* __restrict__ adj, const float* __restrict__ s_lT,
           const float* __restrict__ s_rT, const _Float16* __restrict__ gT,
           const float* __restrict__ ZinvT, float* __restrict__ h_agg)
{
    int tile = blockIdx.x;
    int head = threadIdx.x >> 5;
    int lane = threadIdx.x & 31;
    int i0   = tile * 16;
    int lrow = lane & 15;                 // A row / B column within tile
    int i    = i0 + lrow;
    int kb   = (lane < 16) ? 0 : 16;     // A K-base for this lane half
    int kcol = (lane < 16) ? 0 : 16;     // B K-base for this lane half

    float sl = s_lT[head * NN + i];
    const int*      arow = adj  + (size_t)i * NN;
    const float*    srh  = s_rT + head * NN;
    const _Float16* gTh  = gT   + (size_t)(head * FDIM) * NN;

    v8f c0 = {}, c1 = {}, c2 = {}, c3 = {};

    for (int j0 = 0; j0 < NN; j0 += 32) {
        // adjacency + s_r: one contiguous 16-element segment per lane, vectorized
        union { int4 q[4]; int s[16]; } AD;
        AD.q[0] = *(const int4*)(arow + j0 + kb);
        AD.q[1] = *(const int4*)(arow + j0 + kb + 4);
        AD.q[2] = *(const int4*)(arow + j0 + kb + 8);
        AD.q[3] = *(const int4*)(arow + j0 + kb + 12);
        union { float4 q[4]; float s[16]; } SR;
        SR.q[0] = *(const float4*)(srh + j0 + kb);
        SR.q[1] = *(const float4*)(srh + j0 + kb + 4);
        SR.q[2] = *(const float4*)(srh + j0 + kb + 8);
        SR.q[3] = *(const float4*)(srh + j0 + kb + 12);

        union { v16h v; _Float16 e[16]; } A;
#pragma unroll
        for (int e = 0; e < 16; ++e) {
            // adj is exactly 0 or 1 -> apply mask as a multiply (branchless)
            float p = (float)AD.s[e] * __expf(lrelu(sl + SR.s[e]));
            A.e[e] = (_Float16)p;
        }

        const _Float16* bp = gTh + (size_t)lrow * NN + j0 + kcol;
        v16h b0 = *(const v16h*)(bp);
        v16h b1 = *(const v16h*)(bp + (size_t)16 * NN);
        v16h b2 = *(const v16h*)(bp + (size_t)32 * NN);
        v16h b3 = *(const v16h*)(bp + (size_t)48 * NN);

        c0 = __builtin_amdgcn_wmma_f32_16x16x32_f16(false, A.v, false, b0, (short)0, c0, false, false);
        c1 = __builtin_amdgcn_wmma_f32_16x16x32_f16(false, A.v, false, b1, (short)0, c1, false, false);
        c2 = __builtin_amdgcn_wmma_f32_16x16x32_f16(false, A.v, false, b2, (short)0, c2, false, false);
        c3 = __builtin_amdgcn_wmma_f32_16x16x32_f16(false, A.v, false, b3, (short)0, c3, false, false);
    }

    // epilogue: divide by Z and store (C layout: row = i0 + v + (lane>=16?8:0), col = lane%16)
    const float* zinvh = ZinvT + head * NN;
    int mb = (lane < 16) ? 0 : 8;
#pragma unroll
    for (int v = 0; v < 8; ++v) {
        int row = i0 + mb + v;
        float z = zinvh[row];
        float* o = h_agg + (size_t)row * ONF + head * FDIM + lrow;
        o[0]  = c0[v] * z;
        o[16] = c1[v] * z;
        o[32] = c2[v] * z;
        o[48] = c3[v] * z;
    }
}

// ---------------------------------------------------------------------------
// Kernel 4: per-node encoder MLP: h2 = relu(h@We1.T+b1)@We2.T+b2
// grid 4096 x 256
// ---------------------------------------------------------------------------
__global__ void __launch_bounds__(256)
mlp_kernel(const float* __restrict__ h_agg,
           const float* __restrict__ W_e1, const float* __restrict__ b_e1,
           const float* __restrict__ W_e2, const float* __restrict__ b_e2,
           float* __restrict__ h2)
{
    __shared__ float row[HID];
    __shared__ float h1[HID];
    int i = blockIdx.x, t = threadIdx.x;
    row[t] = h_agg[(size_t)i * HID + t];
    __syncthreads();
    const float* w = W_e1 + t * HID;
    float acc = b_e1[t];
#pragma unroll 8
    for (int k = 0; k < HID; ++k) acc += row[k] * w[k];
    h1[t] = fmaxf(acc, 0.f);
    __syncthreads();
    const float* w2 = W_e2 + t * HID;
    float acc2 = b_e2[t];
#pragma unroll 8
    for (int k = 0; k < HID; ++k) acc2 += h1[k] * w2[k];
    h2[(size_t)i * HID + t] = acc2;
}

// ---------------------------------------------------------------------------
// Kernel 5a: deterministic partial column sums over nodes. grid 32 x 256
// ---------------------------------------------------------------------------
__global__ void __launch_bounds__(256)
psum_kernel(const float* __restrict__ h2, float* __restrict__ partial)
{
    int b = blockIdx.x, t = threadIdx.x;
    int r0 = b * (NN / 32);
    float acc = 0.f;
    for (int r = 0; r < NN / 32; ++r) acc += h2[(size_t)(r0 + r) * HID + t];
    partial[b * HID + t] = acc;
}

// ---------------------------------------------------------------------------
// Kernel 5b: finish pooling + decoder. grid 1 x 256
// ---------------------------------------------------------------------------
__global__ void __launch_bounds__(256)
final_kernel(const float* __restrict__ partial,
             const float* __restrict__ W_d1, const float* __restrict__ b_d1,
             const float* __restrict__ W_d2, const float* __restrict__ b_d2,
             float* __restrict__ out)
{
    __shared__ float pooled[HID];
    __shared__ float red[HID];
    int t = threadIdx.x;
    float acc = 0.f;
    for (int b = 0; b < 32; ++b) acc += partial[b * HID + t];
    pooled[t] = acc;
    __syncthreads();
    const float* w = W_d1 + t * HID;
    float a2 = b_d1[t];
#pragma unroll 8
    for (int k = 0; k < HID; ++k) a2 += pooled[k] * w[k];
    red[t] = fmaxf(a2, 0.f) * W_d2[t];
    __syncthreads();
    for (int s = 128; s; s >>= 1) { if (t < s) red[t] += red[t + s]; __syncthreads(); }
    if (t == 0) out[0] = red[0] + b_d2[0];
}

// ---------------------------------------------------------------------------
extern "C" void kernel_launch(void* const* d_in, const int* in_sizes, int n_in,
                              void* d_out, int out_size, void* d_ws, size_t ws_size,
                              hipStream_t stream)
{
    const float* nodes = (const float*)d_in[0];
    const int*   adj   = (const int*)  d_in[1];
    const float* W_emb = (const float*)d_in[2];
    const float* W_att = (const float*)d_in[3];
    const float* W_e1  = (const float*)d_in[4];
    const float* b_e1  = (const float*)d_in[5];
    const float* W_e2  = (const float*)d_in[6];
    const float* b_e2  = (const float*)d_in[7];
    const float* W_d1  = (const float*)d_in[8];
    const float* b_d1  = (const float*)d_in[9];
    const float* W_d2  = (const float*)d_in[10];
    const float* b_d2  = (const float*)d_in[11];
    float* out = (float*)d_out;

    char* ws = (char*)d_ws;
    size_t off = 0;
    auto alloc = [&](size_t bytes) -> void* {
        void* p = ws + off;
        off += (bytes + 255) & ~(size_t)255;
        return p;
    };
    float*    g       = (float*)   alloc((size_t)NN * ONF * 4);   // 4 MB
    _Float16* gT      = (_Float16*)alloc((size_t)ONF * NN * 2);   // 2 MB
    float*    s_lT    = (float*)   alloc((size_t)HEADS * NN * 4);
    float*    s_rT    = (float*)   alloc((size_t)HEADS * NN * 4);
    float*    ZinvT   = (float*)   alloc((size_t)HEADS * NN * 4);
    float*    h_agg   = (float*)   alloc((size_t)NN * ONF * 4);   // 4 MB
    float*    h2      = (float*)   alloc((size_t)NN * HID * 4);   // 4 MB
    float*    partial = (float*)   alloc((size_t)32 * HID * 4);

    embed_kernel<<<NN, 256, 0, stream>>>(nodes, W_emb, W_att, g, gT, s_lT, s_rT);
    zsum_kernel<<<NN / 8, 256, 0, stream>>>(adj, s_lT, s_rT, ZinvT);
    agg_kernel<<<NN / 16, 128, 0, stream>>>(adj, s_lT, s_rT, gT, ZinvT, h_agg);
    mlp_kernel<<<NN, 256, 0, stream>>>(h_agg, W_e1, b_e1, W_e2, b_e2, h2);
    psum_kernel<<<32, 256, 0, stream>>>(h2, partial);
    final_kernel<<<1, 256, 0, stream>>>(partial, W_d1, b_d1, W_d2, b_d2, out);
}
